// Neighbor_Matcher_19387482374778
// MI455X (gfx1250) — matrix operand order repaired and checked by reading, not compiled, not run on hardware
//
#include <hip/hip_runtime.h>
#include <hip/hip_bf16.h>

#define NENT  200000
#define PADID 199999
#define DIM   256
#define NBR   64
#define NK    21
#define PROJN 200
#define NS    5
#define NQ    1024
#define ROWP  (DIM + 8)   // padded LDS row (halves): 528B stride -> conflict-free WMMA operand loads

typedef __attribute__((ext_vector_type(16))) _Float16 v16h;
typedef __attribute__((ext_vector_type(8)))  float    v8f;

union H16 { v16h h; uint4 u[2]; };

// ---------------- Kernel 1: t = (ent[tail] - ent[head]) @ attn_w^T + attn_b ----------------
__global__ __launch_bounds__(DIM) void nm_t_kernel(
    const int* __restrict__ heads, const int* __restrict__ tails,
    const float* __restrict__ ent_emb, const float* __restrict__ attn_w,
    const float* __restrict__ attn_b, float* __restrict__ t_out)
{
    __shared__ float diff[DIM];
    const int row = blockIdx.x;
    const int d   = threadIdx.x;
    const int h = heads[row], t = tails[row];
    diff[d] = ent_emb[(size_t)t * DIM + d] - ent_emb[(size_t)h * DIM + d];
    __syncthreads();
    float acc = attn_b[d];
    const float* wrow = attn_w + (size_t)d * DIM;   // t[d] = sum_e diff[e]*W[d][e]
#pragma unroll 8
    for (int e = 0; e < DIM; ++e) acc = fmaf(diff[e], wrow[e], acc);
    t_out[(size_t)row * DIM + d] = acc;
}

// --------- Kernel 2: per (row, n): att = dot(t_row, ent[nb]);  e_out = f16(l2norm(rel[nb])) ---------
__global__ __launch_bounds__(DIM) void nm_gather_kernel(
    const int* __restrict__ tails, const float* __restrict__ t_all,
    const int* __restrict__ edge_data, const float* __restrict__ ent_emb,
    const float* __restrict__ rel_emb,
    float* __restrict__ att_out, _Float16* __restrict__ e_out)
{
    const int n   = blockIdx.x;
    const int row = blockIdx.y;
    const int d   = threadIdx.x;
    const int tail = tails[row];
    const int nb = edge_data[(((size_t)tail) * NBR + n) * 2 + 1];
    const float td = t_all[(size_t)row * DIM + d];
    const float e  = ent_emb[(size_t)nb * DIM + d];
    const float r  = rel_emb[(size_t)nb * DIM + d];
    float dot = td * e;
    float n2  = r * r;
#pragma unroll
    for (int m = 16; m >= 1; m >>= 1) {
        dot += __shfl_xor(dot, m, 32);
        n2  += __shfl_xor(n2,  m, 32);
    }
    __shared__ float sdot[8], sn2[8];
    const int wave = d >> 5, lane = d & 31;
    if (lane == 0) { sdot[wave] = dot; sn2[wave] = n2; }
    __syncthreads();
    float D0 = 0.f, Nr = 0.f;
#pragma unroll
    for (int w = 0; w < 8; ++w) { D0 += sdot[w]; Nr += sn2[w]; }
    const float inv = 1.0f / fmaxf(sqrtf(Nr), 1e-12f);
    e_out[(((size_t)row) * NBR + n) * DIM + d] = (_Float16)(r * inv);
    if (d == 0) att_out[(size_t)row * NBR + n] = D0;
}

// ---------------- Kernel 3: fused WMMA sim-GEMM + scale + maxpool + RBF-log + MLP + max_s ----------------
__global__ __launch_bounds__(128) void nm_main_kernel(
    const _Float16* __restrict__ q_e, const _Float16* __restrict__ s_e,
    const float* __restrict__ q_att, const float* __restrict__ s_att,
    const int* __restrict__ edge_data, const int* __restrict__ q_tails,
    const float* __restrict__ d1_w, const float* __restrict__ d1_b,
    const float* __restrict__ d2_w, const float* __restrict__ d2_b,
    float* __restrict__ out)
{
    extern __shared__ __align__(16) char smem[];
    _Float16* lq = (_Float16*)smem;            // [64][ROWP] f16
    _Float16* ls = lq + NBR * ROWP;            // [64][ROWP] f16
    __shared__ float pool[NBR];
    __shared__ float maskf[NBR];
    __shared__ float tenc[24];
    __shared__ float zred[4];

    const int q    = blockIdx.x;
    const int tid  = threadIdx.x;
    const int wave = tid >> 5;
    const int lane = tid & 31;
    const int hi   = lane >> 4;   // half-wave id
    const int ln   = lane & 15;
    const int rowbase = wave * 16;

    // Stage q_e[q] (64x256 f16) into padded LDS (32 x uint4 per row)
    {
        const uint4* src = (const uint4*)(q_e + (size_t)q * NBR * DIM);
        for (int i = tid; i < NBR * (DIM / 8); i += 128) {
            const int r = i >> 5, c = i & 31;
            *(uint4*)(lq + (size_t)r * ROWP + c * 8) = src[(size_t)r * (DIM / 8) + c];
        }
    }
    const int qtail = q_tails[q];
    if (tid < NBR)
        maskf[tid] = (edge_data[(((size_t)qtail) * NBR + tid) * 2 + 1] != PADID) ? 1.0f : 0.0f;

    float qa[8];
#pragma unroll
    for (int r = 0; r < 8; ++r)
        qa[r] = q_att[(size_t)q * NBR + rowbase + hi * 8 + r];

    __syncthreads();

    float best = -1.0f;   // sigmoid outputs are in (0,1)

    for (int s = 0; s < NS; ++s) {
        {   // stage s_e[s] into padded LDS
            const uint4* src = (const uint4*)(s_e + (size_t)s * NBR * DIM);
            for (int i = tid; i < NBR * (DIM / 8); i += 128) {
                const int r = i >> 5, c = i & 31;
                *(uint4*)(ls + (size_t)r * ROWP + c * 8) = src[(size_t)r * (DIM / 8) + c];
            }
        }
        __syncthreads();

        // 16(q rows) x 64(s cols) = 4 tiles, K=256 in 8 steps of 32
        v8f acc[4] = {};
#pragma unroll
        for (int kt = 0; kt < DIM / 32; ++kt) {
            // A operand (16x32 f16): lane holds row M=ln; k chunks [hi*8, hi*8+8) and +16
            H16 ua;
            const _Float16* ap = lq + (size_t)(rowbase + ln) * ROWP + kt * 32 + hi * 8;
            ua.u[0] = *(const uint4*)ap;
            ua.u[1] = *(const uint4*)(ap + 16);
            const v16h a = ua.h;
#pragma unroll
            for (int jt = 0; jt < 4; ++jt) {
                // B operand (32x16 f16): lane holds col N=ln; k chunks [hi*8, hi*8+8) and +16
                H16 ub;
                const _Float16* bp = ls + (size_t)(jt * 16 + ln) * ROWP + kt * 32 + hi * 8;
                ub.u[0] = *(const uint4*)bp;
                ub.u[1] = *(const uint4*)(bp + 16);
                acc[jt] = __builtin_amdgcn_wmma_f32_16x16x32_f16(
                    false, a, false, ub.h, (short)0, acc[jt], false, false);
            }
        }

        // scale by q_att (row) * s_att (col), max over support columns
        float rmax[8];
#pragma unroll
        for (int r = 0; r < 8; ++r) rmax[r] = -3.4e38f;
#pragma unroll
        for (int jt = 0; jt < 4; ++jt) {
            const float sa = s_att[(size_t)s * NBR + jt * 16 + ln];
#pragma unroll
            for (int r = 0; r < 8; ++r)
                rmax[r] = fmaxf(rmax[r], acc[jt][r] * sa * qa[r]);
        }
        // reduce max across the 16 lanes of each half-wave (cols)
#pragma unroll
        for (int m = 8; m >= 1; m >>= 1)
#pragma unroll
            for (int r = 0; r < 8; ++r)
                rmax[r] = fmaxf(rmax[r], __shfl_xor(rmax[r], m, 16));
        if (ln == 0) {
#pragma unroll
            for (int r = 0; r < 8; ++r)
                pool[rowbase + hi * 8 + r] = rmax[r];   // D row = r + 8*hi
        }
        __syncthreads();

        // RBF kernel encode: log(clip(exp(v),1e-30)) == max(v, ln 1e-30); sum over query nbrs
        if (tid < NK) {
            const float mu  = (tid == 0) ? 1.0f : (0.975f - 0.05f * (float)(tid - 1));
            const float is2 = (tid == 0) ? 500000.0f : 50.0f;   // 1/(2*sigma^2)
            float a = 0.0f;
            for (int i = 0; i < NBR; ++i) {
                const float dd = pool[i] - mu;
                float v = -dd * dd * is2;
                v = fmaxf(v, -69.077552789821368f);  // ln(1e-30)
                a += v * maskf[i];
            }
            tenc[tid] = a * 0.01f;
        }
        __syncthreads();

        // MLP: h = tenc @ d1_w^T + d1_b ; z = h @ d2_w + d2_b ; score = sigmoid(z)
        float zp = 0.0f;
        for (int p = tid; p < PROJN; p += 128) {
            float h = d1_b[p];
#pragma unroll
            for (int k = 0; k < NK; ++k) h = fmaf(tenc[k], d1_w[p * NK + k], h);
            zp += h * d2_w[p];
        }
#pragma unroll
        for (int m = 16; m >= 1; m >>= 1) zp += __shfl_xor(zp, m, 32);
        if (lane == 0) zred[wave] = zp;
        __syncthreads();
        if (tid == 0) {
            const float z = zred[0] + zred[1] + zred[2] + zred[3] + d2_b[0];
            best = fmaxf(best, 1.0f / (1.0f + expf(-z)));
        }
        __syncthreads();   // protect LDS before next s iteration
    }
    if (tid == 0) out[q] = best;
}

extern "C" void kernel_launch(void* const* d_in, const int* in_sizes, int n_in,
                              void* d_out, int out_size, void* d_ws, size_t ws_size,
                              hipStream_t stream) {
    (void)in_sizes; (void)n_in; (void)out_size; (void)ws_size;
    const int*   support_heads = (const int*)d_in[0];
    const int*   query_heads   = (const int*)d_in[1];
    const int*   support_tails = (const int*)d_in[2];
    const int*   query_tails   = (const int*)d_in[3];
    const int*   edge_data     = (const int*)d_in[4];
    const float* rel_emb = (const float*)d_in[6];
    const float* ent_emb = (const float*)d_in[7];
    const float* attn_w  = (const float*)d_in[8];
    const float* attn_b  = (const float*)d_in[9];
    const float* d1_w = (const float*)d_in[10];
    const float* d1_b = (const float*)d_in[11];
    const float* d2_w = (const float*)d_in[12];
    const float* d2_b = (const float*)d_in[13];
    float* out = (float*)d_out;

    // workspace carve-up (~35 MB)
    char* ws = (char*)d_ws;
    _Float16* q_e  = (_Float16*)ws;  ws += (size_t)NQ * NBR * DIM * sizeof(_Float16);
    _Float16* s_e  = (_Float16*)ws;  ws += (size_t)NS * NBR * DIM * sizeof(_Float16);
    float*    q_at = (float*)ws;     ws += (size_t)NQ * NBR * sizeof(float);
    float*    s_at = (float*)ws;     ws += (size_t)NS * NBR * sizeof(float);
    float*    t_q  = (float*)ws;     ws += (size_t)NQ * DIM * sizeof(float);
    float*    t_s  = (float*)ws;     ws += (size_t)NS * DIM * sizeof(float);

    nm_t_kernel<<<NQ, DIM, 0, stream>>>(query_heads,   query_tails,   ent_emb, attn_w, attn_b, t_q);
    nm_t_kernel<<<NS, DIM, 0, stream>>>(support_heads, support_tails, ent_emb, attn_w, attn_b, t_s);
    nm_gather_kernel<<<dim3(NBR, NQ), DIM, 0, stream>>>(query_tails,   t_q, edge_data, ent_emb, rel_emb, q_at, q_e);
    nm_gather_kernel<<<dim3(NBR, NS), DIM, 0, stream>>>(support_tails, t_s, edge_data, ent_emb, rel_emb, s_at, s_e);

    const size_t smem = (size_t)2 * NBR * ROWP * sizeof(_Float16);  // 67,584 B dynamic LDS
    hipFuncSetAttribute((const void*)nm_main_kernel,
                        hipFuncAttributeMaxDynamicSharedMemorySize, (int)smem);
    nm_main_kernel<<<NQ, 128, smem, stream>>>(q_e, s_e, q_at, s_at, edge_data, query_tails,
                                              d1_w, d1_b, d2_w, d2_b, out);
}
